// ClusterLoss_51273319579921
// MI455X (gfx1250) — matrix operand order repaired and checked by reading, not compiled, not run on hardware
//
#include <hip/hip_runtime.h>
#include <hip/hip_bf16.h>
#include <math.h>

typedef __attribute__((ext_vector_type(16))) _Float16 v16h_t;
typedef __attribute__((ext_vector_type(8)))  _Float16 v8h_t;
typedef __attribute__((ext_vector_type(8)))  float    v8f_t;

#define ALPHA_C  6.0f
#define BETA_C   2.0f
#define ANN_R_C  3.0f
#define ANN_RR_C 5.0f
#define EPS_C    1e-5f

// ---------------------------------------------------------------------------
// Zero-init scratch accumulators (rowacc[N*4] + annulus scalar).
// ---------------------------------------------------------------------------
__global__ void cl_zero_kernel(float* __restrict__ p, int n) {
    int i = blockIdx.x * blockDim.x + threadIdx.x;
    if (i < n) p[i] = 0.0f;
}

// ---------------------------------------------------------------------------
// Per-row: cf = feat - center, split into f16 hi/lo, row sum of squares,
// annulus-loss contribution (atomic into annacc[0]).
// One block (256 threads) per row.
// ---------------------------------------------------------------------------
__global__ void cl_center_split_kernel(const float* __restrict__ feat,
                                       const float* __restrict__ center,
                                       _Float16* __restrict__ cf_hi,
                                       _Float16* __restrict__ cf_lo,
                                       float* __restrict__ sq,
                                       float* __restrict__ annacc,
                                       int D) {
    const int row  = blockIdx.x;
    const int tid  = threadIdx.x;
    const size_t base = (size_t)row * (size_t)D;

    float s = 0.0f;
    for (int c = tid; c < D; c += blockDim.x) {
        float v = feat[base + c] - center[c];
        _Float16 h = (_Float16)v;
        _Float16 l = (_Float16)(v - (float)h);
        cf_hi[base + c] = h;
        cf_lo[base + c] = l;
        s += v * v;
    }
    // wave32 reduce
    for (int m = 16; m >= 1; m >>= 1) s += __shfl_xor(s, m, 32);
    __shared__ float red[8];
    const int lane = tid & 31, w = tid >> 5;
    if (lane == 0) red[w] = s;
    __syncthreads();
    if (tid == 0) {
        float t = 0.0f;
        const int nw = blockDim.x >> 5;
        for (int k = 0; k < nw; ++k) t += red[k];
        sq[row] = t;
        float ac = sqrtf(fmaxf(t, 1e-12f));
        float contrib = fmaxf(ANN_R_C - ac, 0.0f) + fmaxf(ac - ANN_RR_C, 0.0f);
        atomicAdd(annacc, contrib);
    }
}

// A-fragment loader (16x32 f16, row-major source): elements 0..7 at p,
// elements 8..15 at p+8 (contiguous K, matching the B fragment).
__device__ __forceinline__ v16h_t cl_load_afrag(const _Float16* p) {
    v8h_t a0 = *(const v8h_t*)p;
    v8h_t a1 = *(const v8h_t*)(p + 8);
    return __builtin_shufflevector(a0, a1,
        0,1,2,3,4,5,6,7,8,9,10,11,12,13,14,15);
}

// ---------------------------------------------------------------------------
// One wave32 per 32x32 output region (2x2 register-blocked 16x16 WMMA tiles)
// of G = cf * cf^T via split-f16 WMMA:
//   G ~= hi*hi^T + hi*lo^T + lo*hi^T
// => 12x v_wmma_f32_16x16x32_f16 per K=32 step on 8 fragments (24 FLOP per
// L2 byte vs 6 for the unblocked version).
// Fused epilogue: dm = 1 - sqrt(sq_i + sq_j - 2G), upper-triangular label
// masks, exp weighting, per-row partial sums -> atomicAdd into rowacc[i*4..].
// 32x32 regions strictly below the diagonal are skipped (mask is j > i).
// ---------------------------------------------------------------------------
__global__ void __launch_bounds__(32)
cl_tile_kernel(const _Float16* __restrict__ cf_hi,
               const _Float16* __restrict__ cf_lo,
               const float* __restrict__ sq,
               const int* __restrict__ labels,
               float* __restrict__ rowacc,
               int N, int D) {
    const int ti = blockIdx.y;
    const int tj = blockIdx.x;
    if (tj < ti) return;              // uniform per-wave exit; EXEC stays all-1s

    const int lane = threadIdx.x;     // 0..31 (wave32)
    const int i0 = ti * 32;
    const int j0 = tj * 32;

    // A fragment (16x32 f16): lane holds row base + (lane&15);
    //   elements 0..15 = K (lane>=16 ? 16 : 0) + 0..15 (contiguous)
    // B fragment (32x16 f16): lane holds column base + (lane&15);
    //   elements 0..15 = K (lane>=16 ? 16 : 0) + 0..15 (contiguous)
    const int r15 = lane & 15;
    const int kbA = (lane >> 4) * 16;
    const int kbB = (lane >> 4) * 16;

    const _Float16* aHi[2];
    const _Float16* aLo[2];
    const _Float16* bHi[2];
    const _Float16* bLo[2];
    aHi[0] = cf_hi + (size_t)(i0 + r15) * D + kbA;
    aHi[1] = aHi[0] + (size_t)16 * D;
    aLo[0] = cf_lo + (size_t)(i0 + r15) * D + kbA;
    aLo[1] = aLo[0] + (size_t)16 * D;
    bHi[0] = cf_hi + (size_t)(j0 + r15) * D + kbB;
    bHi[1] = bHi[0] + (size_t)16 * D;
    bLo[0] = cf_lo + (size_t)(j0 + r15) * D + kbB;
    bLo[1] = bLo[0] + (size_t)16 * D;

    v8f_t acc[2][2] = {};
    for (int k0 = 0; k0 < D; k0 += 32) {
        v16h_t aH[2], aL[2], bH[2], bL[2];
#pragma unroll
        for (int t = 0; t < 2; ++t) {
            aH[t] = cl_load_afrag(aHi[t] + k0);
            aL[t] = cl_load_afrag(aLo[t] + k0);
            bH[t] = *(const v16h_t*)(bHi[t] + k0);
            bL[t] = *(const v16h_t*)(bLo[t] + k0);
        }
        __builtin_prefetch(aHi[0] + k0 + 512, 0, 1);
        __builtin_prefetch(bHi[0] + k0 + 512, 0, 1);

#pragma unroll
        for (int ii = 0; ii < 2; ++ii) {
#pragma unroll
            for (int jj = 0; jj < 2; ++jj) {
                acc[ii][jj] = __builtin_amdgcn_wmma_f32_16x16x32_f16(
                    false, aH[ii], false, bH[jj], (short)0, acc[ii][jj], false, false);
                acc[ii][jj] = __builtin_amdgcn_wmma_f32_16x16x32_f16(
                    false, aH[ii], false, bL[jj], (short)0, acc[ii][jj], false, false);
                acc[ii][jj] = __builtin_amdgcn_wmma_f32_16x16x32_f16(
                    false, aL[ii], false, bH[jj], (short)0, acc[ii][jj], false, false);
            }
        }
    }

    // Epilogue. C/D layout: VGPR v -> row M = v + (lane>=16 ? 8 : 0),
    // column N = lane & 15.
    const int mrow = (lane >> 4) << 3;

#pragma unroll
    for (int jj = 0; jj < 2; ++jj) {
        const int j    = j0 + jj * 16 + r15;
        const float sqj = sq[j];
        const int  lblj = labels[j];
#pragma unroll
        for (int ii = 0; ii < 2; ++ii) {
            for (int v = 0; v < 8; ++v) {
                const int i = i0 + ii * 16 + v + mrow;
                float g    = acc[ii][jj][v];
                float d2   = sq[i] + sqj - 2.0f * g;
                float dist = sqrtf(fmaxf(d2, 1e-12f));
                float dm   = 1.0f - dist;

                const bool valid = (j > i);
                const bool same  = (labels[i] == lblj);

                float ap = 0.0f, np = 0.0f, mw = 0.0f, w = 0.0f;
                if (valid && same) {
                    ap = fmaxf(dm - BETA_C, 0.0f);
                    np = 1.0f;
                }
                if (valid && !same && (dm < ALPHA_C)) {
                    float margin = ALPHA_C - dm;
                    w  = __expf(margin);
                    mw = margin * w;
                }
                // reduce across the 16-lane column group (xor masks stay in-half)
                for (int s = 1; s < 16; s <<= 1) {
                    ap += __shfl_xor(ap, s, 32);
                    np += __shfl_xor(np, s, 32);
                    mw += __shfl_xor(mw, s, 32);
                    w  += __shfl_xor(w,  s, 32);
                }
                if (r15 == 0) {
                    float* r = rowacc + (size_t)i * 4;
                    atomicAdd(r + 0, ap);
                    atomicAdd(r + 1, np);
                    atomicAdd(r + 2, mw);
                    atomicAdd(r + 3, w);
                }
            }
        }
    }
}

// ---------------------------------------------------------------------------
// Final scalar: mean_i( ap_i/(np_i+eps) + mw_i/(w_i+eps) ) + annulus/N
// ---------------------------------------------------------------------------
__global__ void cl_finalize_kernel(const float* __restrict__ rowacc,
                                   const float* __restrict__ annacc,
                                   float* __restrict__ out, int N) {
    const int tid = threadIdx.x;
    float s = 0.0f;
    for (int i = tid; i < N; i += blockDim.x) {
        const float* r = rowacc + (size_t)i * 4;
        s += r[0] / (r[1] + EPS_C) + r[2] / (r[3] + EPS_C);
    }
    for (int m = 16; m >= 1; m >>= 1) s += __shfl_xor(s, m, 32);
    __shared__ float red[8];
    const int lane = tid & 31, w = tid >> 5;
    if (lane == 0) red[w] = s;
    __syncthreads();
    if (tid == 0) {
        float t = 0.0f;
        const int nw = blockDim.x >> 5;
        for (int k = 0; k < nw; ++k) t += red[k];
        out[0] = t / (float)N + annacc[0] / (float)N;
    }
}

// ---------------------------------------------------------------------------
extern "C" void kernel_launch(void* const* d_in, const int* in_sizes, int n_in,
                              void* d_out, int out_size, void* d_ws, size_t ws_size,
                              hipStream_t stream) {
    const float* feat   = (const float*)d_in[0];
    const float* center = (const float*)d_in[1];
    const int*   labels = (const int*)d_in[2];
    float*       out    = (float*)d_out;

    const int D = in_sizes[1];            // center is [1, D]
    const int N = in_sizes[0] / D;        // feat is [N, D]

    // Workspace layout (bytes):
    //   [0 .. N*D*2)            cf_hi  (f16)
    //   [N*D*2 .. N*D*4)        cf_lo  (f16)
    //   then: sq[N], rowacc[N*4], annacc[1]   (floats)
    char* ws = (char*)d_ws;
    _Float16* cf_hi  = (_Float16*)ws;
    _Float16* cf_lo  = (_Float16*)(ws + (size_t)N * D * 2);
    float*    sq     = (float*)(ws + (size_t)N * D * 4);
    float*    rowacc = sq + N;
    float*    annacc = rowacc + (size_t)N * 4;

    // 1) zero rowacc + annulus accumulator
    const int nz = N * 4 + 1;
    cl_zero_kernel<<<(nz + 255) / 256, 256, 0, stream>>>(rowacc, nz);

    // 2) center, f16 hi/lo split, row norms, annulus term
    cl_center_split_kernel<<<N, 256, 0, stream>>>(feat, center, cf_hi, cf_lo,
                                                  sq, annacc, D);

    // 3) WMMA Gram 32x32 regions + fused masked reductions (upper tiles only)
    dim3 grid(N / 32, N / 32);
    cl_tile_kernel<<<grid, 32, 0, stream>>>(cf_hi, cf_lo, sq, labels,
                                            rowacc, N, D);

    // 4) final scalar
    cl_finalize_kernel<<<1, 256, 0, stream>>>(rowacc, annacc, out, N);
}
